// DU_SB_66795331387677
// MI455X (gfx1250) — matrix operand order repaired and checked by reading, not compiled, not run on hardware
//
#include <hip/hip_runtime.h>

typedef _Float16 h16;
typedef __attribute__((ext_vector_type(8)))  _Float16 v8h;
typedef __attribute__((ext_vector_type(16))) _Float16 v16h;
typedef __attribute__((ext_vector_type(8)))  float    v8f;

#define N_DIM   4096
#define B_DIM   512
#define T_STEPS 100
#define BM      128
#define BN      128
#define KT      32
#define LDSS    40      // padded LDS row stride in halves (80 bytes)

union V16 { v16h v; v8h h[2]; };
union F8  { float f[8]; float4 v4[2]; };

__device__ __forceinline__ float sigm(float v) { return 1.0f / (1.0f + __expf(-v)); }

// Async DMA: global -> LDS, 16 bytes per lane, tracked by ASYNCcnt.
__device__ __forceinline__ void async_b128(unsigned lds_off, unsigned long long gaddr)
{
    asm volatile("global_load_async_to_lds_b128 %0, %1, off"
                 :: "v"(lds_off), "v"(gaddr)
                 : "memory");
}
__device__ __forceinline__ void wait_async0()
{
    asm volatile("s_wait_asynccnt 0x0" ::: "memory");
}

// ---------------------------------------------------------------------------
// J fp32 -> f16 conversion + per-row partial sum of squares (for ||J||_F)
// ---------------------------------------------------------------------------
__global__ __launch_bounds__(256)
void sb_convJ(const float* __restrict__ J, h16* __restrict__ Jh,
              float* __restrict__ partial)
{
    __shared__ float red[256];
    const int tid = threadIdx.x;
    const size_t row = (size_t)blockIdx.x * N_DIM;
    float s = 0.0f;
#pragma unroll
    for (int i = 0; i < 16; ++i) {
        size_t o = row + tid + (size_t)i * 256;
        float v = J[o];
        Jh[o] = (h16)v;
        s += v * v;
    }
    red[tid] = s;
    __syncthreads();
#pragma unroll
    for (int st = 128; st > 0; st >>= 1) {
        if (tid < st) red[tid] += red[tid + st];
        __syncthreads();
    }
    if (tid == 0) partial[blockIdx.x] = red[0];
}

// ---------------------------------------------------------------------------
// finalize scale = eta * 0.5*sqrt(N-1)/||J||_F   (1 block, deterministic)
// ---------------------------------------------------------------------------
__global__ __launch_bounds__(256)
void sb_scale(const float* __restrict__ partial, const float* __restrict__ eta,
              float* __restrict__ scale)
{
    __shared__ float red[256];
    const int tid = threadIdx.x;
    float s = 0.0f;
    for (int i = tid; i < N_DIM; i += 256) s += partial[i];
    red[tid] = s;
    __syncthreads();
#pragma unroll
    for (int st = 128; st > 0; st >>= 1) {
        if (tid < st) red[tid] += red[tid + st];
        __syncthreads();
    }
    if (tid == 0) {
        float c0 = 0.5f * sqrtf((float)(N_DIM - 1)) / sqrtf(red[0]);
        *scale = eta[0] * c0;
    }
}

// ---------------------------------------------------------------------------
// state init: x = 0.02*(x0-0.5), y = 0.02*(y0-0.5), stored transposed [B][N]
// ---------------------------------------------------------------------------
__global__ __launch_bounds__(256)
void sb_init(const float* __restrict__ x0, const float* __restrict__ y0,
             float* __restrict__ Xf, float* __restrict__ Yf, h16* __restrict__ Xh)
{
    const int idx = blockIdx.x * 256 + threadIdx.x;   // idx = b*4096 + m
    const int m = idx & (N_DIM - 1);
    const int b = idx >> 12;
    const size_t in = (size_t)m * B_DIM + b;          // inputs are [N][B]
    float x = 0.02f * (x0[in] - 0.5f);
    float y = 0.02f * (y0[in] - 0.5f);
    Xf[idx] = x;
    Yf[idx] = y;
    Xh[idx] = (h16)x;
}

// ---------------------------------------------------------------------------
// One SB step:  G = J @ x  (WMMA f16->f32), fused elementwise update epilogue.
// Async-DMA double-buffered LDS pipeline (ASYNCcnt), 8 WMMAs per K-tile/wave.
// ---------------------------------------------------------------------------
__global__ __launch_bounds__(256)
void sb_step(const h16* __restrict__ Jh,
             const h16* __restrict__ Xin,
             h16* __restrict__ Xout,
             float* __restrict__ Xf,
             float* __restrict__ Yf,
             const float* __restrict__ hvec,
             const float* __restrict__ scale_p,
             const float* __restrict__ delta_t,
             float ak)
{
    __shared__ h16 Jl[2][BM * LDSS];
    __shared__ h16 Xl[2][BN * LDSS];

    const int tid  = threadIdx.x;
    const int wid  = tid >> 5;
    const int lane = tid & 31;
    const int l16  = lane & 15;
    const int hi   = lane >> 4;            // 0 | 1

    const int m_blk = blockIdx.y * BM;
    const int n_blk = blockIdx.x * BN;

    const int wm = (wid >> 2) * 64;        // wave tile: 64 (M) x 32 (N)
    const int wn = (wid & 3) * 32;

    // global->LDS: 512 16B segments per tile, 2 per thread per tile
    const int r0 = tid >> 2,         p0 = tid & 3;
    const int r1 = (tid + 256) >> 2, p1 = (tid + 256) & 3;

    // global byte addresses (advance 64B per K-tile)
    unsigned long long aJ0 = (unsigned long long)(Jh  + (size_t)(m_blk + r0) * N_DIM + p0 * 8);
    unsigned long long aJ1 = (unsigned long long)(Jh  + (size_t)(m_blk + r1) * N_DIM + p1 * 8);
    unsigned long long aX0 = (unsigned long long)(Xin + (size_t)(n_blk + r0) * N_DIM + p0 * 8);
    unsigned long long aX1 = (unsigned long long)(Xin + (size_t)(n_blk + r1) * N_DIM + p1 * 8);

    // LDS byte offsets for both buffers (LDS offset = low 32 bits of generic addr)
    const unsigned lJ0[2] = { (unsigned)(unsigned long long)&Jl[0][r0 * LDSS + p0 * 8],
                              (unsigned)(unsigned long long)&Jl[1][r0 * LDSS + p0 * 8] };
    const unsigned lJ1[2] = { (unsigned)(unsigned long long)&Jl[0][r1 * LDSS + p1 * 8],
                              (unsigned)(unsigned long long)&Jl[1][r1 * LDSS + p1 * 8] };
    const unsigned lX0[2] = { (unsigned)(unsigned long long)&Xl[0][r0 * LDSS + p0 * 8],
                              (unsigned)(unsigned long long)&Xl[1][r0 * LDSS + p0 * 8] };
    const unsigned lX1[2] = { (unsigned)(unsigned long long)&Xl[0][r1 * LDSS + p1 * 8],
                              (unsigned)(unsigned long long)&Xl[1][r1 * LDSS + p1 * 8] };

    v8f acc[4][2];
#pragma unroll
    for (int a = 0; a < 4; ++a)
#pragma unroll
        for (int b = 0; b < 2; ++b)
            acc[a][b] = (v8f){0.f, 0.f, 0.f, 0.f, 0.f, 0.f, 0.f, 0.f};

    const int NKI = N_DIM / KT;            // 128

    // prologue: DMA K-tile 0 into buffer 0
    async_b128(lJ0[0], aJ0);
    async_b128(lJ1[0], aJ1);
    async_b128(lX0[0], aX0);
    async_b128(lX1[0], aX1);
    wait_async0();
    __syncthreads();

    const int kA = hi ? 8 : 0;             // A frag: K {0..7,16..23} | {8..15,24..31}
    const int kB = hi ? 8 : 0;             // B frag: K {0..7,16..23} | {8..15,24..31}

#pragma unroll 1
    for (int kt = 0; kt < NKI; ++kt) {
        const int cur = kt & 1;
        const bool more = (kt + 1 < NKI);
        if (more) {                        // DMA next tile into other buffer
            const unsigned long long kb = (unsigned long long)(kt + 1) * (KT * sizeof(h16));
            const int nxt = cur ^ 1;
            async_b128(lJ0[nxt], aJ0 + kb);
            async_b128(lJ1[nxt], aJ1 + kb);
            async_b128(lX0[nxt], aX0 + kb);
            async_b128(lX1[nxt], aX1 + kb);
        }

        const h16* Jb = &Jl[cur][0];
        const h16* Xb = &Xl[cur][0];

        V16 af[4], bf[2];
#pragma unroll
        for (int a = 0; a < 4; ++a) {
            const int row = (wm + a * 16 + l16) * LDSS;
            af[a].h[0] = *(const v8h*)&Jb[row + kA];
            af[a].h[1] = *(const v8h*)&Jb[row + kA + 16];
        }
#pragma unroll
        for (int b = 0; b < 2; ++b) {
            const int col = (wn + b * 16 + l16) * LDSS;
            bf[b].h[0] = *(const v8h*)&Xb[col + kB];
            bf[b].h[1] = *(const v8h*)&Xb[col + kB + 16];
        }
#pragma unroll
        for (int a = 0; a < 4; ++a)
#pragma unroll
            for (int b = 0; b < 2; ++b)
                acc[a][b] = __builtin_amdgcn_wmma_f32_16x16x32_f16(
                    false, af[a].v, false, bf[b].v, (short)0, acc[a][b], false, false);

        if (more) {
            wait_async0();                 // own DMAs into next buffer done
            __syncthreads();               // everyone's DMAs visible
        }
    }

    // ---------------- fused elementwise epilogue -----------------
    const float dk   = *delta_t;           // uniform scalar loads
    const float sc   = *scale_p;
    const float omak = 1.0f - ak;

#pragma unroll
    for (int a = 0; a < 4; ++a) {
        const int m0 = m_blk + wm + a * 16 + hi * 8;   // lane owns M = m0..m0+7
        F8 H;
        H.v4[0] = *(const float4*)&hvec[m0];
        H.v4[1] = *(const float4*)&hvec[m0 + 4];
#pragma unroll
        for (int b = 0; b < 2; ++b) {
            const int n = n_blk + wn + b * 16 + l16;
            const size_t base = (size_t)n * N_DIM + m0;    // contiguous 8 floats
            F8 X, Y;
            X.v4[0] = *(const float4*)&Xf[base];
            X.v4[1] = *(const float4*)&Xf[base + 4];
            Y.v4[0] = *(const float4*)&Yf[base];
            Y.v4[1] = *(const float4*)&Yf[base + 4];
            v8h xh;
#pragma unroll
            for (int r = 0; r < 8; ++r) {
                float g = acc[a][b][r];                    // (J@x)[m0+r, n]
                float y = Y.f[r] + dk * (-omak * X.f[r] + sc * (g + H.f[r]));
                float x = X.f[r] + dk * y;
                float vp = x + 1.0f, vm = x - 1.0f;
                float xn = vp * sigm(vp) - vm * sigm(vm) - 1.0f;   // phi_s
                float ps = sigm(100.0f * (fabsf(xn) - 1.01f));     // psi_s
                y *= (1.0f - ps);
                X.f[r] = xn;
                Y.f[r] = y;
                xh[r] = (h16)xn;
            }
            *(float4*)&Xf[base]     = X.v4[0];
            *(float4*)&Xf[base + 4] = X.v4[1];
            *(float4*)&Yf[base]     = Y.v4[0];
            *(float4*)&Yf[base + 4] = Y.v4[1];
            *(v8h*)&Xout[base]      = xh;
        }
    }
}

// ---------------------------------------------------------------------------
extern "C" void kernel_launch(void* const* d_in, const int* in_sizes, int n_in,
                              void* d_out, int out_size, void* d_ws, size_t ws_size,
                              hipStream_t stream)
{
    const float* J     = (const float*)d_in[0];
    const float* hvec  = (const float*)d_in[1];
    const float* x0    = (const float*)d_in[2];
    const float* y0    = (const float*)d_in[3];
    const float* delta = (const float*)d_in[4];
    const float* eta   = (const float*)d_in[5];

    float* Xf = (float*)d_out;                 // fp32 x in [B][N] == final x.T

    char* ws = (char*)d_ws;
    h16*   Jh      = (h16*)ws;                                  // 32 MB
    h16*   Xh0     = (h16*)(ws + ((size_t)32 << 20));           // 4 MB
    h16*   Xh1     = (h16*)(ws + ((size_t)36 << 20));           // 4 MB
    float* Yf      = (float*)(ws + ((size_t)40 << 20));         // 8 MB
    float* partial = (float*)(ws + ((size_t)48 << 20));         // 16 KB
    float* scale   = (float*)(ws + ((size_t)48 << 20) + 16384); // 4 B
    h16*   Xh[2]   = {Xh0, Xh1};

    sb_convJ<<<N_DIM, 256, 0, stream>>>(J, Jh, partial);
    sb_scale<<<1, 256, 0, stream>>>(partial, eta, scale);
    sb_init<<<(N_DIM * B_DIM) / 256, 256, 0, stream>>>(x0, y0, Xf, Yf, Xh[0]);

    dim3 grid(B_DIM / BN, N_DIM / BM);         // (4, 32)
    for (int t = 0; t < T_STEPS; ++t) {
        float ak = (float)t / (float)(T_STEPS - 1);
        sb_step<<<grid, 256, 0, stream>>>(Jh, Xh[t & 1], Xh[(t + 1) & 1],
                                          Xf, Yf, hvec, scale, delta + t, ak);
    }
}
